// Net_7559142441492
// MI455X (gfx1250) — matrix operand order repaired and checked by reading, not compiled, not run on hardware
//
#include <hip/hip_runtime.h>

#define G   512
#define NPG 100
#define DD  128
#define EPG 1600
#define NN  (G * NPG)
#define EE  (G * EPG)

typedef __bf16 v16bf __attribute__((ext_vector_type(16)));
typedef float  v8f   __attribute__((ext_vector_type(8)));
typedef unsigned int v4u __attribute__((ext_vector_type(4)));

union Frag { v16bf v; v4u q[2]; };

__device__ __forceinline__ unsigned short f2bf(float f) {
    union { float f; unsigned u; } x; x.f = f;
    unsigned u = x.u;
    unsigned r = u + 0x7FFFu + ((u >> 16) & 1u);   // round-to-nearest-even
    return (unsigned short)(r >> 16);
}

// ---- LDS layout (bytes) for the fused layer kernel ----
// agg   f32[12800]   @ 0        (51200 B)  -- layer output H
// deg   f32[128]     @ 51200    (512 B)    -- [120] holds ||pw||^2
// Abf   bf16[14336]  @ 51712    (28672 B)  -- mean, padded to 112 rows
// Xbf   bf16[14336]  @ 80384    (28672 B)  -- x, padded to 112 rows
// pwS   f32[128]     @ 109056
// sc    f32[128]     @ 109568
// tsc   f32[128]     @ 110080
// keep  i32[128]     @ 110592
// epack u16[1600]    @ 111104   (3200 B)   -- packed (src|dst<<8) or 0xFFFF
// offv  i32[128]     @ 114304   (512 B)    -- CSR row offsets
// csrc  u16[1600]    @ 114816   (3200 B)   -- CSR source-node lists
#define SMEM_BYTES 118016

__global__ void init_kernel(float* hAcc, unsigned char* mask) {
    int i = blockIdx.x * blockDim.x + threadIdx.x;
    if (i < G * DD) hAcc[i] = 0.0f;
    if (i < NN)     mask[i] = 1;
}

__global__ void embed_kernel(const int* __restrict__ idx,
                             const float* __restrict__ emb,
                             float* __restrict__ x) {
    int i = blockIdx.x * blockDim.x + threadIdx.x;
    if (i < NN * DD) {
        int n = i >> 7, d = i & 127;
        x[i] = emb[idx[n] * DD + d];
    }
}

// Convert one 128x128 f32 weight W[k][n] into bf16 transposed Wt[n][k]
__global__ void prep_w_kernel(const float* __restrict__ W,
                              unsigned short* __restrict__ Wt) {
    int i = blockIdx.x * blockDim.x + threadIdx.x;   // 16384 threads
    int n = i >> 7, k = i & 127;
    Wt[n * DD + k] = f2bf(W[k * DD + n]);
}

// Fused per-graph layer: CSR mean-aggregate -> WMMA(mean@Wl + x@Wr + b) -> relu
//                        -> topk score/select -> write pooled x, mask, gap
__global__ void layer_kernel(const int* __restrict__ src,
                             const int* __restrict__ dst,
                             float* __restrict__ xBuf,
                             unsigned char* __restrict__ mask,
                             float* __restrict__ hAcc,
                             const unsigned short* __restrict__ WlT,
                             const unsigned short* __restrict__ WrT,
                             const float* __restrict__ bl,
                             const float* __restrict__ pw,
                             int Kkeep, float invK) {
    extern __shared__ char smem[];
    float*          agg   = (float*)(smem);                 // H output
    float*          deg   = (float*)(smem + 51200);
    unsigned short* Abf   = (unsigned short*)(smem + 51712);
    unsigned short* Xbf   = (unsigned short*)(smem + 80384);
    float*          pwS   = (float*)(smem + 109056);
    float*          sc    = (float*)(smem + 109568);
    float*          tsc   = (float*)(smem + 110080);
    int*            keep  = (int*)(smem + 110592);
    unsigned short* epack = (unsigned short*)(smem + 111104);
    int*            offv  = (int*)(smem + 114304);
    unsigned short* csrc  = (unsigned short*)(smem + 114816);

    const int g = blockIdx.x;
    const int tid = threadIdx.x;
    const int gNode = g * NPG;
    const int eBase = g * EPG;

    // ---- phase 1: stage packed valid edges into LDS; stage pw ----
    for (int i = tid; i < EPG; i += 256) {
        int s = src[eBase + i], t = dst[eBase + i];
        bool valid = mask[s] && mask[t];
        epack[i] = valid ? (unsigned short)((s - gNode) | ((t - gNode) << 8))
                         : (unsigned short)0xFFFF;
    }
    if (tid < 128) pwS[tid] = pw[tid];
    __syncthreads();

    // ---- phase 2a: per-node in-degree (parallel over 100 nodes) ----
    if (tid < NPG) {
        int c = 0;
        for (int e = 0; e < EPG; ++e) {
            unsigned up = epack[e];
            if (up != 0xFFFFu && (int)(up >> 8) == tid) ++c;
        }
        offv[tid] = c;
        deg[tid]  = (float)c;
    }
    __syncthreads();

    // ---- phase 2b: exclusive scan of counts; pw norm ----
    if (tid == 0) {
        int run = 0;
        for (int n = 0; n < NPG; ++n) { int c = offv[n]; offv[n] = run; run += c; }
    } else if (tid == 32) {
        float s = 0.0f;
        for (int d = 0; d < DD; ++d) s += pwS[d] * pwS[d];
        deg[120] = s;                       // ||pw||^2 (rows 100+ unused)
    }
    __syncthreads();

    // ---- phase 2c: fill CSR lists in original edge order (deterministic) ----
    if (tid < NPG) {
        int p = offv[tid];
        for (int e = 0; e < EPG; ++e) {
            unsigned up = epack[e];
            if (up != 0xFFFFu && (int)(up >> 8) == tid)
                csrc[p++] = (unsigned short)(up & 0xFFu);
        }
    }
    __syncthreads();

    // ---- phase 3: parallel gather-mean into Abf; stage Xbf; zero pads ----
    {
        const int wave = tid >> 5, lane = tid & 31;
        // 400 warp-tasks: (node, 32-dim chunk)
        for (int task = wave; task < NPG * 4; task += 8) {
            int n = task >> 2;
            int d = ((task & 3) << 5) + lane;
            int base = offv[n];
            int cnt  = (int)deg[n];
            float s = 0.0f;
            for (int j = 0; j < cnt; ++j) {
                int sLoc = csrc[base + j];
                s += xBuf[(gNode + sLoc) * DD + d];
            }
            float dv = cnt > 0 ? (float)cnt : 1.0f;
            Abf[n * DD + d] = f2bf(s / dv);
        }
    }
    for (int i = tid; i < 112 * DD; i += 256) {
        int row = i >> 7;
        Xbf[i] = (row < NPG) ? f2bf(xBuf[gNode * DD + i]) : (unsigned short)0;
        if (row >= NPG) Abf[i] = 0;
    }
    __syncthreads();

    // ---- phase 4: WMMA  H = relu(mean@Wl + x@Wr + bl) ----
    {
        const int wave   = tid >> 5;         // 8 waves -> 8 column tiles
        const int lane   = tid & 31;
        const int laneHi = (lane >= 16) ? 1 : 0;
        const int ln     = lane & 15;
        const int nc     = wave * 16 + ln;   // output column
        const int aoff   = laneHi ? 8 : 0;   // A layout: hi lanes hold K+8
        const int boff   = laneHi ? 8 : 0;   // B layout: hi lanes hold K+8
        for (int mtile = 0; mtile < 7; ++mtile) {
            const int rowA = mtile * 16 + ln;
            v8f c = {0.f, 0.f, 0.f, 0.f, 0.f, 0.f, 0.f, 0.f};
            for (int kb = 0; kb < DD; kb += 32) {
                Frag a, b;
                const unsigned short* ap = Abf + rowA * DD + kb + aoff;
                a.q[0] = *(const v4u*)(ap);
                a.q[1] = *(const v4u*)(ap + 16);
                const unsigned short* bp = WlT + nc * DD + kb + boff;
                b.q[0] = *(const v4u*)(bp);
                b.q[1] = *(const v4u*)(bp + 16);
                c = __builtin_amdgcn_wmma_f32_16x16x32_bf16(
                        false, a.v, false, b.v, (short)0, c, false, false);
            }
            for (int kb = 0; kb < DD; kb += 32) {
                Frag a, b;
                const unsigned short* ap = Xbf + rowA * DD + kb + aoff;
                a.q[0] = *(const v4u*)(ap);
                a.q[1] = *(const v4u*)(ap + 16);
                const unsigned short* bp = WrT + nc * DD + kb + boff;
                b.q[0] = *(const v4u*)(bp);
                b.q[1] = *(const v4u*)(bp + 16);
                c = __builtin_amdgcn_wmma_f32_16x16x32_bf16(
                        false, a.v, false, b.v, (short)0, c, false, false);
            }
            const float bias = bl[nc];
            for (int r = 0; r < 8; ++r) {
                int row = mtile * 16 + r + laneHi * 8;
                float v = c[r] + bias;
                v = v > 0.0f ? v : 0.0f;
                if (row < NPG) agg[row * DD + nc] = v;
            }
        }
    }
    __syncthreads();

    // ---- phase 5: projection scores ----
    if (tid < NPG) {
        const float* hrow = agg + tid * DD;
        float dot = 0.0f;
        for (int d = 0; d < DD; ++d) dot += hrow[d] * pwS[d];
        sc[tid] = mask[gNode + tid] ? dot * rsqrtf(deg[120]) : -1e9f;
    }
    __syncthreads();

    // ---- phase 6: top-K by rank counting (stable, index tie-break) ----
    if (tid < NPG) {
        float s = sc[tid];
        int rank = 0;
        for (int j = 0; j < NPG; ++j) {
            float sj = sc[j];
            if (sj > s || (sj == s && j < tid)) ++rank;
        }
        keep[tid] = (rank < Kkeep) ? 1 : 0;
        tsc[tid]  = tanhf(s);
    }
    __syncthreads();

    // ---- phase 7: write pooled x (in-place), mask, and gap contribution ----
    for (int i = tid; i < NPG * DD; i += 256) {
        int row = i >> 7;
        float v = keep[row] ? agg[i] * tsc[row] : 0.0f;
        xBuf[gNode * DD + i] = v;
    }
    if (tid < NPG) mask[gNode + tid] = (unsigned char)keep[tid];
    if (tid < DD) {
        float s = 0.0f;
        for (int row = 0; row < NPG; ++row)
            if (keep[row]) s += agg[row * DD + tid] * tsc[row];
        hAcc[g * DD + tid] += s * invK;
    }
}

// Tiny MLP head: h -> relu(lin1) -> relu(lin2) -> sigmoid(lin3)
__global__ void head_kernel(const float* __restrict__ hAcc,
                            const float* __restrict__ w1, const float* __restrict__ b1,
                            const float* __restrict__ w2, const float* __restrict__ b2,
                            const float* __restrict__ w3, const float* __restrict__ b3,
                            float* __restrict__ out) {
    __shared__ float hS[128], y1[128], y2[64];
    const int g = blockIdx.x, t = threadIdx.x;
    hS[t] = hAcc[g * 128 + t];
    __syncthreads();
    {
        float s = b1[t];
        for (int d = 0; d < 128; ++d) s += hS[d] * w1[d * 128 + t];
        y1[t] = s > 0.0f ? s : 0.0f;
    }
    __syncthreads();
    if (t < 64) {
        float s = b2[t];
        for (int d = 0; d < 128; ++d) s += y1[d] * w2[d * 64 + t];
        y2[t] = s > 0.0f ? s : 0.0f;
    }
    __syncthreads();
    if (t == 0) {
        float s = b3[0];
        for (int d = 0; d < 64; ++d) s += y2[d] * w3[d];
        out[g] = 1.0f / (1.0f + expf(-s));
    }
}

extern "C" void kernel_launch(void* const* d_in, const int* in_sizes, int n_in,
                              void* d_out, int out_size, void* d_ws, size_t ws_size,
                              hipStream_t stream) {
    const int*   x_idx = (const int*)d_in[0];
    const int*   ei    = (const int*)d_in[1];
    const int*   srcI  = ei;
    const int*   dstI  = ei + EE;
    const float* emb   = (const float*)d_in[2];
    const float* w_l[3] = {(const float*)d_in[3],  (const float*)d_in[7],  (const float*)d_in[11]};
    const float* b_l[3] = {(const float*)d_in[4],  (const float*)d_in[8],  (const float*)d_in[12]};
    const float* w_r[3] = {(const float*)d_in[5],  (const float*)d_in[9],  (const float*)d_in[13]};
    const float* pw[3]  = {(const float*)d_in[6],  (const float*)d_in[10], (const float*)d_in[14]};
    const float* lin1_w = (const float*)d_in[15];
    const float* lin1_b = (const float*)d_in[16];
    const float* lin2_w = (const float*)d_in[17];
    const float* lin2_b = (const float*)d_in[18];
    const float* lin3_w = (const float*)d_in[19];
    const float* lin3_b = (const float*)d_in[20];
    float* out = (float*)d_out;

    // workspace carve-up (all 16B aligned)
    char* ws = (char*)d_ws;
    float* xBuf = (float*)ws;                                  // NN*DD f32
    float* hAcc = xBuf + (size_t)NN * DD;                      // G*DD f32
    unsigned char* mask = (unsigned char*)(hAcc + G * DD);     // NN bytes
    unsigned short* wbf = (unsigned short*)(mask + NN);        // 6 * 128*128 bf16

    init_kernel<<<256, 256, 0, stream>>>(hAcc, mask);
    embed_kernel<<<(NN * DD + 255) / 256, 256, 0, stream>>>(x_idx, emb, xBuf);
    for (int l = 0; l < 3; ++l) {
        prep_w_kernel<<<64, 256, 0, stream>>>(w_l[l], wbf + (size_t)(2 * l) * 16384);
        prep_w_kernel<<<64, 256, 0, stream>>>(w_r[l], wbf + (size_t)(2 * l + 1) * 16384);
    }
    const int Ks[3] = {80, 64, 52};
    for (int l = 0; l < 3; ++l) {
        layer_kernel<<<G, 256, SMEM_BYTES, stream>>>(
            srcI, dstI, xBuf, mask, hAcc,
            wbf + (size_t)(2 * l) * 16384, wbf + (size_t)(2 * l + 1) * 16384,
            b_l[l], pw[l], Ks[l], 1.0f / (float)Ks[l]);
    }
    head_kernel<<<G, 128, 0, stream>>>(hAcc, lin1_w, lin1_b, lin2_w, lin2_b,
                                       lin3_w, lin3_b, out);
}
